// ContentAwareUpsample_82025285419163
// MI455X (gfx1250) — matrix operand order repaired and checked by reading, not compiled, not run on hardware
//
#include <hip/hip_runtime.h>
#include <hip/hip_bf16.h>

typedef __attribute__((ext_vector_type(16))) _Float16 v16h;
typedef __attribute__((ext_vector_type(8)))  _Float16 v8h;
typedef __attribute__((ext_vector_type(8)))  float    v8f;

#define B_   8
#define C_   64
#define H_   64
#define W_   64
#define HW_  (H_ * W_)
#define KP_  2304          // C_*9*2*2
#define OUT_H (2 * H_)
#define OUT_W (2 * W_)

// ---------------------------------------------------------------------------
// CDNA5 async global->LDS copy (ASYNCcnt-tracked, per-lane 16B) + wait.
// Inline asm is portable across the ROCm7.2 / amdgpu-toolchain builtin split.
// Generic pointers to __shared__ carry the LDS byte offset in their low 32 bits.
// ---------------------------------------------------------------------------
__device__ __forceinline__ void async_b128_g2l(_Float16* ldsp, const _Float16* gp) {
    unsigned int loff = (unsigned int)(unsigned long long)ldsp;
    asm volatile("global_load_async_to_lds_b128 %0, %1, off"
                 :: "v"(loff), "v"(gp)
                 : "memory");
}
__device__ __forceinline__ void wait_async0() {
    asm volatile("s_wait_asynccnt 0x0" ::: "memory");
}

// ---------------------------------------------------------------------------
// Layout / precision prep kernels
// ---------------------------------------------------------------------------

// ctx (B,C,H,W) fp32 -> [b][h*W+w][c] f16 (channel contiguous per pixel)
__global__ void k_prep_act(const float* __restrict__ src, _Float16* __restrict__ dst, int n) {
    int i = blockIdx.x * blockDim.x + threadIdx.x;
    if (i >= n) return;
    int hw = i % HW_;
    int t  = i / HW_;
    int c  = t % C_;
    int b  = t / C_;
    dst[((size_t)b * HW_ + hw) * C_ + c] = (_Float16)src[i];
}

// conv weight OIHW (cout,cin,3,3) fp32 -> [cout][tap*cin + ci] f16
__global__ void k_prep_w(const float* __restrict__ src, _Float16* __restrict__ dst,
                         int cin, int n) {
    int i = blockIdx.x * blockDim.x + threadIdx.x;
    if (i >= n) return;
    int tap = i % 9;
    int t   = i / 9;
    int ci  = t % cin;
    int co  = t / cin;
    dst[(size_t)co * (9 * cin) + tap * cin + ci] = (_Float16)src[i];
}

// kp weight (2304,32,1,1) fp32 -> [oc][ic] f16 (already that layout)
__global__ void k_prep_kp(const float* __restrict__ src, _Float16* __restrict__ dst, int n) {
    int i = blockIdx.x * blockDim.x + threadIdx.x;
    if (i >= n) return;
    dst[i] = (_Float16)src[i];
}

// ---------------------------------------------------------------------------
// Fragment loaders (f16 WMMA A operand in this file's contiguous k order, wave32)
// ---------------------------------------------------------------------------

// A-matrix 16x32 f16: lane L holds row M = L%16.
// lanes 0-15:  halves 0-15 = K 0..15
// lanes 16-31: halves 0-15 = K 16..31
__device__ __forceinline__ v16h load_a_frag(const _Float16* __restrict__ wrow,
                                            int kbase, int lane) {
    int off = (lane < 16) ? 0 : 16;
    const v8h* p0 = (const v8h*)(wrow + kbase + off);
    const v8h* p1 = (const v8h*)(wrow + kbase + 8 + off);
    v16h a;
#pragma unroll
    for (int i = 0; i < 8; ++i) { a[i] = (*p0)[i]; a[8 + i] = (*p1)[i]; }
    return a;
}

// ---------------------------------------------------------------------------
// Shared activation staging for the 3x3 convs:
// rows h-1..h+1 of [b][hw][64] f16 into LDS with a zero halo column each side.
// sact[r][cc][ci], cc = w+1 for w in -1..64. Async copies for in-range rows,
// explicit zero stores for out-of-range rows and halo columns.
// ---------------------------------------------------------------------------
#define SACT_COLS 66
__device__ __forceinline__ void stage_rows(_Float16 (*sact)[SACT_COLS][64],
                                           const _Float16* __restrict__ act,
                                           int b, int h, int tid) {
    // 3 rows * 64 pixels * 8 chunks(16B) = 1536 chunks; 256 threads -> 6 each
#pragma unroll
    for (int it = 0; it < 6; ++it) {
        int idx   = tid + it * 256;
        int r     = idx >> 9;            // 0..2
        int chunk = idx & 511;
        int pix   = chunk >> 3;
        int sub   = chunk & 7;
        int hh    = h - 1 + r;
        _Float16* lp = &sact[r][pix + 1][sub * 8];
        if (hh >= 0 && hh < H_) {
            const _Float16* gp = act + ((size_t)b * HW_ + hh * W_ + pix) * 64 + sub * 8;
            async_b128_g2l(lp, gp);
        } else {
            *(v8h*)lp = (v8h){};
        }
    }
    // halo columns cc=0 and cc=65: 3 rows * 2 cols * 8 chunks = 48 chunks
    if (tid < 48) {
        int r   = tid / 16;
        int t   = tid % 16;
        int cc  = (t < 8) ? 0 : (SACT_COLS - 1);
        int sub = t & 7;
        *(v8h*)&sact[r][cc][sub * 8] = (v8h){};
    }
    wait_async0();
    __syncthreads();
}

// ---------------------------------------------------------------------------
// conv1: 3x3, 64->64, pad=1 (zero), +bias, +PReLU. Implicit GEMM, f16 WMMA.
// One block per (batch,row). 8 waves; wave owns 2 cout-tiles x 1 pixel-tile.
// B fragments come from LDS (halo makes the tap loop branchless).
// ---------------------------------------------------------------------------
__global__ __launch_bounds__(256) void k_conv1(
    const _Float16* __restrict__ act,   // [B][HW][64]
    const _Float16* __restrict__ wgt,   // [64][9*64]
    const float* __restrict__ bias,     // [64]
    const float* __restrict__ alpha,    // [64]
    _Float16* __restrict__ out)         // [B][HW][64]
{
    __shared__ __align__(16) _Float16 sact[3][SACT_COLS][64];   // 25.3 KB
    int h = blockIdx.x % H_;
    int b = blockIdx.x / H_;
    stage_rows(sact, act, b, h, threadIdx.x);

    int lane = threadIdx.x & 31;
    int wave = threadIdx.x >> 5;
    int nt   = wave & 3;            // pixel tile (16 pixels)
    int mt0  = (wave >> 2) * 2;     // cout tiles mt0, mt0+1
    int col  = lane & 15;
    int wp   = nt * 16 + col;       // this lane's pixel column (B frag / D frag)
    int koff = (lane < 16) ? 0 : 16;
    int roff = (lane < 16) ? 0 : 8;

    int cb0 = mt0 * 16 + roff;      // D-frag cout base for acc0
    int cb1 = cb0 + 16;
    v8f acc0, acc1;
#pragma unroll
    for (int r = 0; r < 8; ++r) { acc0[r] = bias[cb0 + r]; acc1[r] = bias[cb1 + r]; }

    const _Float16* w0row = wgt + (size_t)(mt0 * 16 + col) * (9 * 64);
    const _Float16* w1row = w0row + (size_t)16 * (9 * 64);

#pragma unroll
    for (int tap = 0; tap < 9; ++tap) {
        int dy = tap / 3;
        int dx = tap % 3;
        const _Float16* bp = &sact[dy][wp + dx][koff];   // wp+dx-1 (+1 halo shift)
#pragma unroll
        for (int ck = 0; ck < 2; ++ck) {
            v16h bf = *(const v16h*)(bp + ck * 32);      // 2x ds_load_b128
            v16h a0 = load_a_frag(w0row, tap * 64 + ck * 32, lane);
            v16h a1 = load_a_frag(w1row, tap * 64 + ck * 32, lane);
            acc0 = __builtin_amdgcn_wmma_f32_16x16x32_f16(false, a0, false, bf,
                                                          (short)0, acc0, false, false);
            acc1 = __builtin_amdgcn_wmma_f32_16x16x32_f16(false, a1, false, bf,
                                                          (short)0, acc1, false, false);
        }
    }

    // PReLU + f16 store: each lane holds one pixel, 8 consecutive couts per acc.
    _Float16* o = out + ((size_t)b * HW_ + h * W_ + wp) * 64 + cb0;
    v8h s0, s1;
#pragma unroll
    for (int r = 0; r < 8; ++r) {
        float v0 = acc0[r]; v0 = v0 > 0.f ? v0 : alpha[cb0 + r] * v0; s0[r] = (_Float16)v0;
        float v1 = acc1[r]; v1 = v1 > 0.f ? v1 : alpha[cb1 + r] * v1; s1[r] = (_Float16)v1;
    }
    *(v8h*)o        = s0;
    *(v8h*)(o + 16) = s1;
}

// ---------------------------------------------------------------------------
// conv2: 3x3, 64->32, pad=1, +bias, +PReLU. 8 waves: 2 mt x 4 nt = 1 tile/wave.
// ---------------------------------------------------------------------------
__global__ __launch_bounds__(256) void k_conv2(
    const _Float16* __restrict__ act,   // [B][HW][64]
    const _Float16* __restrict__ wgt,   // [32][9*64]
    const float* __restrict__ bias,     // [32]
    const float* __restrict__ alpha,    // [32]
    _Float16* __restrict__ out)         // [B][HW][32]
{
    __shared__ __align__(16) _Float16 sact[3][SACT_COLS][64];
    int h = blockIdx.x % H_;
    int b = blockIdx.x / H_;
    stage_rows(sact, act, b, h, threadIdx.x);

    int lane = threadIdx.x & 31;
    int wave = threadIdx.x >> 5;
    int nt   = wave & 3;
    int mt   = wave >> 2;           // 0..1
    int col  = lane & 15;
    int wp   = nt * 16 + col;
    int koff = (lane < 16) ? 0 : 16;
    int roff = (lane < 16) ? 0 : 8;

    int cb = mt * 16 + roff;
    v8f acc;
#pragma unroll
    for (int r = 0; r < 8; ++r) acc[r] = bias[cb + r];

    const _Float16* wrow = wgt + (size_t)(mt * 16 + col) * (9 * 64);

#pragma unroll
    for (int tap = 0; tap < 9; ++tap) {
        int dy = tap / 3;
        int dx = tap % 3;
        const _Float16* bp = &sact[dy][wp + dx][koff];
#pragma unroll
        for (int ck = 0; ck < 2; ++ck) {
            v16h bf = *(const v16h*)(bp + ck * 32);
            v16h a = load_a_frag(wrow, tap * 64 + ck * 32, lane);
            acc = __builtin_amdgcn_wmma_f32_16x16x32_f16(false, a, false, bf,
                                                         (short)0, acc, false, false);
        }
    }

    _Float16* o = out + ((size_t)b * HW_ + h * W_ + wp) * 32 + cb;
    v8h s;
#pragma unroll
    for (int r = 0; r < 8; ++r) {
        float v = acc[r]; v = v > 0.f ? v : alpha[cb + r] * v; s[r] = (_Float16)v;
    }
    *(v8h*)o = s;
}

// ---------------------------------------------------------------------------
// Fused kp 1x1 conv (32->2304, K=32 == one f16 WMMA) + softmax(9) + upsample.
// One block per (batch,row,16-pixel tile). Mask tile (2304x16 f32) lives only
// in LDS (144 KB of the WGP's 320 KB) — never touches HBM (saves ~600 MB RT).
// ---------------------------------------------------------------------------
__global__ __launch_bounds__(256) void k_upsample(
    const _Float16* __restrict__ f2,    // [B][HW][32]
    const _Float16* __restrict__ kpw,   // [2304][32]
    const float* __restrict__ kpb,      // [2304]
    const float* __restrict__ x_low,    // [B][64][64][64] fp32
    float* __restrict__ out)            // [B][64][128][128] fp32
{
    extern __shared__ float lds[];      // 2304*16 floats
    int wt = blockIdx.x & 3;
    int t  = blockIdx.x >> 2;
    int h  = t % H_;
    int b  = t / H_;
    int w0 = wt * 16;

    // Prefetch the x_low window phase 2 will need (overlaps the 18 WMMAs).
    {
        int c = threadIdx.x >> 2;
        int r = threadIdx.x & 3;
        if (r < 3) {
            int hh = h - 1 + r;
            hh = hh < 0 ? 0 : (hh > H_ - 1 ? H_ - 1 : hh);
            int wl = w0 - 1; wl = wl < 0 ? 0 : wl;
            const float* pf = x_low + ((size_t)b * C_ + c) * HW_ + hh * W_ + wl;
            __builtin_prefetch(pf, 0, 3);        // global_prefetch_b8
            __builtin_prefetch(pf + 16, 0, 3);
        }
    }

    int lane = threadIdx.x & 31;
    int wave = threadIdx.x >> 5;
    int col  = lane & 15;
    int koff = (lane < 16) ? 0 : 16;
    int roff = (lane < 16) ? 0 : 8;

    // B fragment: 16 pixels x 32 channels, loaded once per wave.
    const _Float16* bp = f2 + ((size_t)b * HW_ + h * W_ + w0 + col) * 32 + koff;
    v16h bf = *(const v16h*)bp;

    // 144 oc-tiles over 8 waves -> 18 WMMAs per wave, results to LDS.
    for (int i = 0; i < 18; ++i) {
        int ocb = (wave * 18 + i) * 16;
        const _Float16* wrow = kpw + (size_t)(ocb + col) * 32;
        v16h a = load_a_frag(wrow, 0, lane);
        v8f acc;
#pragma unroll
        for (int r = 0; r < 8; ++r) acc[r] = kpb[ocb + roff + r];
        acc = __builtin_amdgcn_wmma_f32_16x16x32_f16(false, a, false, bf,
                                                     (short)0, acc, false, false);
#pragma unroll
        for (int r = 0; r < 8; ++r) lds[(ocb + roff + r) * 16 + col] = acc[r];
    }
    __syncthreads();

    // Phase 2: softmax over 9 taps + edge-padded neighbor blend.
    // tasks = c(64) * p(2) * q(2) * px(16) = 4096 -> 16 per thread.
#pragma unroll 1
    for (int it = 0; it < 16; ++it) {
        int task = threadIdx.x + it * 256;
        int px = task & 15;
        int q  = (task >> 4) & 1;
        int p  = (task >> 5) & 1;
        int c  = task >> 6;
        int w  = w0 + px;
        int ob = c * 36 + p * 2 + q;        // oc = c*36 + k*4 + p*2 + q

        float m[9];
        float mx = -1e30f;
#pragma unroll
        for (int k = 0; k < 9; ++k) {
            m[k] = lds[(ob + k * 4) * 16 + px];
            mx = fmaxf(mx, m[k]);
        }
        float s = 0.f;
#pragma unroll
        for (int k = 0; k < 9; ++k) { m[k] = __expf(m[k] - mx); s += m[k]; }
        float inv = 1.0f / s;

        const float* xb = x_low + ((size_t)b * C_ + c) * HW_;
        float acc = 0.f;
#pragma unroll
        for (int k = 0; k < 9; ++k) {
            int hh = h + k / 3 - 1;
            int ww = w + k % 3 - 1;
            hh = hh < 0 ? 0 : (hh > H_ - 1 ? H_ - 1 : hh);   // edge padding
            ww = ww < 0 ? 0 : (ww > W_ - 1 ? W_ - 1 : ww);
            acc += m[k] * xb[hh * W_ + ww];
        }
        acc *= inv;

        // Reference does transpose(0,1,5,3,4,2): out[b,c, w*2+q, h*2+p]
        int y = w * 2 + q;
        int x = h * 2 + p;
        out[(((size_t)b * C_ + c) * OUT_H + y) * OUT_W + x] = acc;
    }
}

// ---------------------------------------------------------------------------
extern "C" void kernel_launch(void* const* d_in, const int* in_sizes, int n_in,
                              void* d_out, int out_size, void* d_ws, size_t ws_size,
                              hipStream_t stream) {
    (void)in_sizes; (void)n_in; (void)out_size; (void)ws_size;
    const float* x_low = (const float*)d_in[0];
    const float* ctx   = (const float*)d_in[1];
    const float* c1_w  = (const float*)d_in[2];
    const float* c1_b  = (const float*)d_in[3];
    const float* p1_a  = (const float*)d_in[4];
    const float* c2_w  = (const float*)d_in[5];
    const float* c2_b  = (const float*)d_in[6];
    const float* p2_a  = (const float*)d_in[7];
    const float* kp_w  = (const float*)d_in[8];
    const float* kp_b  = (const float*)d_in[9];

    char* ws = (char*)d_ws;
    _Float16* ctxh = (_Float16*)ws;  ws += (size_t)B_ * HW_ * 64 * 2;
    _Float16* f1h  = (_Float16*)ws;  ws += (size_t)B_ * HW_ * 64 * 2;
    _Float16* f2h  = (_Float16*)ws;  ws += (size_t)B_ * HW_ * 32 * 2;
    _Float16* w1h  = (_Float16*)ws;  ws += (size_t)64 * 576 * 2;
    _Float16* w2h  = (_Float16*)ws;  ws += (size_t)32 * 576 * 2;
    _Float16* kph  = (_Float16*)ws;  ws += (size_t)KP_ * 32 * 2;

    int nAct = B_ * C_ * HW_;
    k_prep_act<<<(nAct + 255) / 256, 256, 0, stream>>>(ctx, ctxh, nAct);
    k_prep_w<<<(64 * 64 * 9 + 255) / 256, 256, 0, stream>>>(c1_w, w1h, 64, 64 * 64 * 9);
    k_prep_w<<<(32 * 64 * 9 + 255) / 256, 256, 0, stream>>>(c2_w, w2h, 64, 32 * 64 * 9);
    k_prep_kp<<<(KP_ * 32 + 255) / 256, 256, 0, stream>>>(kp_w, kph, KP_ * 32);

    k_conv1<<<B_ * H_, 256, 0, stream>>>(ctxh, w1h, c1_b, p1_a, f1h);
    k_conv2<<<B_ * H_, 256, 0, stream>>>(f1h, w2h, c2_b, p2_a, f2h);

    size_t lds_bytes = (size_t)KP_ * 16 * sizeof(float);   // 147456 B < 320 KB WGP LDS
    k_upsample<<<B_ * H_ * 4, 256, lds_bytes, stream>>>(f2h, kph, kp_b, x_low, (float*)d_out);
}
